// TextLSTM_63153199120834
// MI455X (gfx1250) — matrix operand order repaired and checked by reading, not compiled, not run on hardware
//
#include <hip/hip_runtime.h>
#include <hip/hip_bf16.h>

#define SEQ 512
#define EMB 512
#define HID 512
#define NC  32000
#define G4  2048   // 4 * HID

typedef __attribute__((ext_vector_type(16))) _Float16 v16h;
typedef __attribute__((ext_vector_type(8)))  float    v8f;

// ---------------------------------------------------------------------------
// Prep kernels: gather embedding row for batch 255, convert/transpose weights.
// ---------------------------------------------------------------------------
__global__ void __launch_bounds__(256)
prep_embed(const int* __restrict__ X, const float* __restrict__ emb,
           _Float16* __restrict__ Xe) {
    int s = blockIdx.x;                  // 0..511 (timestep)
    int e = threadIdx.x;                 // 0..255
    int tok = X[255 * SEQ + s];
    const float* row = emb + (long)tok * EMB;
    Xe[s * EMB + e]        = (_Float16)row[e];
    Xe[s * EMB + e + 256]  = (_Float16)row[e + 256];
}

// WxT[j][e] = Wx_k[e][h]  with j = k*512 + h  (column-major for WMMA B loads)
__global__ void __launch_bounds__(256)
prep_wxt(const float* __restrict__ Wi, const float* __restrict__ Wf,
         const float* __restrict__ Wg, const float* __restrict__ Wo,
         _Float16* __restrict__ WxT) {
    int k = blockIdx.x >> 9;             // gate
    int e = blockIdx.x & 511;            // input feature
    const float* W = (k == 0) ? Wi : (k == 1) ? Wf : (k == 2) ? Wg : Wo;
    for (int h = threadIdx.x; h < HID; h += 256)
        WxT[(long)(k * HID + h) * EMB + e] = (_Float16)W[e * HID + h];
}

// Straight f32 -> f16 conversion of the recurrent weights, same layout.
__global__ void __launch_bounds__(256)
prep_wh(const float* __restrict__ Wi, const float* __restrict__ Wf,
        const float* __restrict__ Wg, const float* __restrict__ Wo,
        _Float16* __restrict__ Wh) {
    long idx = (long)blockIdx.x * 256 + threadIdx.x;   // 0 .. 4*512*512-1
    int  k   = (int)(idx >> 18);
    long m   = idx & 262143;
    const float* W = (k == 0) ? Wi : (k == 1) ? Wf : (k == 2) ? Wg : Wo;
    Wh[idx] = (_Float16)W[m];
}

// Combined gate biases + zero the inter-WG barrier counters (every launch).
__global__ void __launch_bounds__(256)
prep_misc(const float* __restrict__ bii, const float* __restrict__ bhi,
          const float* __restrict__ bif, const float* __restrict__ bhf,
          const float* __restrict__ big, const float* __restrict__ bhg,
          const float* __restrict__ bio, const float* __restrict__ bho,
          float* __restrict__ bx, unsigned int* __restrict__ counters) {
    int j = blockIdx.x * 256 + threadIdx.x;            // 0..2047
    int k = j >> 9, h = j & 511;
    const float* bi = (k == 0) ? bii : (k == 1) ? bif : (k == 2) ? big : bio;
    const float* bh = (k == 0) ? bhi : (k == 1) ? bhf : (k == 2) ? bhg : bho;
    bx[j] = bi[h] + bh[h];
    if (j < SEQ) counters[j] = 0u;
}

// ---------------------------------------------------------------------------
// K1: input-projection GEMM via WMMA.
//   G[s][j] = sum_e Xe[s][e] * WxT[j][e] + bx[j]     (s: 512, j: 2048, e: 512)
// One wave per 16x16 output tile; K-loop in chunks of 32 (f16 WMMA).
// Fragment packing follows CDNA5 ISA 7.12.2 tables.
// ---------------------------------------------------------------------------
__global__ void __launch_bounds__(128)
gx_wmma(const _Float16* __restrict__ Xe, const _Float16* __restrict__ WxT,
        const float* __restrict__ bx, float* __restrict__ G) {
    int wave = blockIdx.x * 4 + (threadIdx.x >> 5);    // 0..4095
    int lane = threadIdx.x & 31;
    int mt = wave >> 7;                                // 0..31   M-tile
    int nt = wave & 127;                               // 0..127  N-tile
    int lo = lane & 15;
    bool hi = lane >= 16;

    int arow  = mt * 16 + lo;                          // A: M = lane%16
    int akb2  = hi ? 4 : 0;                            // A K-base (uint units)
    int ncol  = nt * 16 + lo;                          // B: N = lane%16
    int bkb2  = hi ? 4 : 0;                            // B K-base (uint units)

    const unsigned int* Arow = (const unsigned int*)(Xe + (long)arow * EMB);
    const unsigned int* Brow = (const unsigned int*)(WxT + (long)ncol * EMB);

    v8f acc = {};
    for (int kb = 0; kb < EMB; kb += 32) {
        union { unsigned int u[8]; v16h v; } A, B;
        int base = kb >> 1;                            // uint index of chunk
        #pragma unroll
        for (int v = 0; v < 4; ++v) {
            A.u[v]     = Arow[base + akb2 + v];        // K = akb + 2v, +1
            A.u[v + 4] = Arow[base + 8 + akb2 + v];    // K = 16 + akb + 2v
        }
        #pragma unroll
        for (int v = 0; v < 8; ++v)
            B.u[v] = Brow[base + (v >> 2) * 8 + bkb2 + (v & 3)];   // same K order as A
        acc = __builtin_amdgcn_wmma_f32_16x16x32_f16(
            false, A.v, false, B.v, (short)0, acc, false, false);
    }

    float b = bx[ncol];
    int rbase = mt * 16 + (hi ? 8 : 0);                // C/D: M = vgpr + hi*8
    #pragma unroll
    for (int v = 0; v < 8; ++v)
        G[(long)(rbase + v) * G4 + ncol] = acc[v] + b;
}

// ---------------------------------------------------------------------------
// K2: serial LSTM recurrence for batch row 255.
// 16 WGs x 256 threads. WG w owns gh columns [128w, 128w+128). Per step:
//   gh[j] = sum_e h[e] * Whf16[k][e][j%512],  j = k*512 + col
// split across 2 half-dots per column, then device-scope barrier, then every
// WG redundantly applies the gate update so h,c stay in its own LDS.
// gh is double-buffered so one barrier per step suffices.
// ---------------------------------------------------------------------------
__global__ void __launch_bounds__(256)
lstm_recurrent(const float* __restrict__ hidden, const float* __restrict__ cell,
               const _Float16* __restrict__ Wh, const float* __restrict__ G,
               float* gh, float* __restrict__ hfinal,
               unsigned int* counters) {
    __shared__ float h_s[HID];
    __shared__ float c_s[HID];
    __shared__ float part[256];

    const int tid = threadIdx.x;
    const int w   = blockIdx.x;                        // 0..15

    h_s[tid]       = hidden[255 * HID + tid];
    h_s[tid + 256] = hidden[255 * HID + tid + 256];
    c_s[tid]       = cell[255 * HID + tid];
    c_s[tid + 256] = cell[255 * HID + tid + 256];
    __syncthreads();

    const int jj   = w * 128 + (tid & 127);            // global column 0..2047
    const int k    = jj >> 9;
    const int col  = jj & 511;
    const int half = tid >> 7;                         // which K half
    const _Float16* Wbase = Wh + (long)k * HID * HID + col;

    for (int t = 0; t < SEQ; ++t) {
        // --- partial dot products ---
        float acc = 0.f;
        const int e0 = half * 256;
        #pragma unroll 8
        for (int e = e0; e < e0 + 256; ++e)
            acc += h_s[e] * (float)Wbase[(long)e * HID];
        part[tid] = acc;
        __syncthreads();

        float* ghb = gh + (t & 1) * G4;
        if (tid < 128)
            ghb[w * 128 + tid] = part[tid] + part[tid + 128];
        __threadfence();
        __syncthreads();

        // --- device-scope barrier across the 16 WGs (step-indexed) ---
        if (tid == 0) {
            __hip_atomic_fetch_add(&counters[t], 1u, __ATOMIC_RELEASE,
                                   __HIP_MEMORY_SCOPE_AGENT);
            while (__hip_atomic_load(&counters[t], __ATOMIC_ACQUIRE,
                                     __HIP_MEMORY_SCOPE_AGENT) < 16u)
                __builtin_amdgcn_s_sleep(1);
        }
        __syncthreads();
        __threadfence();

        // --- redundant elementwise gate update (all 512 units per WG) ---
        const float* Gt = G + (long)t * G4;
        #pragma unroll
        for (int r = 0; r < 2; ++r) {
            int u = tid + r * 256;
            float zi = Gt[u]            + ghb[u];
            float zf = Gt[HID + u]      + ghb[HID + u];
            float zg = Gt[2 * HID + u]  + ghb[2 * HID + u];
            float zo = Gt[3 * HID + u]  + ghb[3 * HID + u];
            float ig = 1.f / (1.f + __expf(-zi));
            float fg = 1.f / (1.f + __expf(-zf));
            float gg = tanhf(zg);
            float og = 1.f / (1.f + __expf(-zo));
            float cn = fg * c_s[u] + ig * gg;
            c_s[u] = cn;
            h_s[u] = og * tanhf(cn);
        }
        __syncthreads();
    }

    if (w == 0) {
        hfinal[tid]       = h_s[tid];
        hfinal[tid + 256] = h_s[tid + 256];
    }
}

// ---------------------------------------------------------------------------
// K3: logits.  out[n] = b_out[n] + sum_e hfinal[e] * W_out[e][n]
// Fully coalesced streaming of the 64 MB W_out (HBM-bound, ~3 us).
// ---------------------------------------------------------------------------
__global__ void __launch_bounds__(256)
logits_kernel(const float* __restrict__ hfinal, const float* __restrict__ Wout,
              const float* __restrict__ bout, float* __restrict__ out) {
    __shared__ float h_s[HID];
    int tid = threadIdx.x;
    h_s[tid]       = hfinal[tid];
    h_s[tid + 256] = hfinal[tid + 256];
    __syncthreads();
    int n = blockIdx.x * 256 + tid;                    // grid is exactly 32000
    float acc = bout[n];
    #pragma unroll 8
    for (int e = 0; e < HID; ++e)
        acc += h_s[e] * Wout[(long)e * NC + n];
    out[n] = acc;
}

// ---------------------------------------------------------------------------
extern "C" void kernel_launch(void* const* d_in, const int* in_sizes, int n_in,
                              void* d_out, int out_size, void* d_ws, size_t ws_size,
                              hipStream_t stream) {
    const int*   X      = (const int*)  d_in[0];
    const float* hidden = (const float*)d_in[1];
    const float* cell   = (const float*)d_in[2];
    const float* emb    = (const float*)d_in[3];
    const float* W_ii   = (const float*)d_in[4];
    const float* b_ii   = (const float*)d_in[5];
    const float* W_hi   = (const float*)d_in[6];
    const float* b_hi   = (const float*)d_in[7];
    const float* W_if   = (const float*)d_in[8];
    const float* b_if   = (const float*)d_in[9];
    const float* W_hf   = (const float*)d_in[10];
    const float* b_hf   = (const float*)d_in[11];
    const float* W_ig   = (const float*)d_in[12];
    const float* b_ig   = (const float*)d_in[13];
    const float* W_hg   = (const float*)d_in[14];
    const float* b_hg   = (const float*)d_in[15];
    const float* W_io   = (const float*)d_in[16];
    const float* b_io   = (const float*)d_in[17];
    const float* W_ho   = (const float*)d_in[18];
    const float* b_ho   = (const float*)d_in[19];
    const float* W_out  = (const float*)d_in[20];
    const float* b_out  = (const float*)d_in[21];
    float* out = (float*)d_out;

    // Workspace carve (256B aligned slices)
    size_t off = 0;
    char* base = (char*)d_ws;
    auto carve = [&](size_t bytes) -> void* {
        void* p = base + off;
        off += (bytes + 255) & ~(size_t)255;
        return p;
    };
    _Float16*     Xe   = (_Float16*)    carve((size_t)SEQ * EMB * 2);       // 512 KB
    _Float16*     WxT  = (_Float16*)    carve((size_t)G4 * EMB * 2);        // 2 MB
    _Float16*     Whc  = (_Float16*)    carve((size_t)4 * HID * HID * 2);   // 2 MB
    float*        bx   = (float*)       carve((size_t)G4 * 4);
    float*        G    = (float*)       carve((size_t)SEQ * G4 * 4);        // 4 MB
    float*        gh   = (float*)       carve((size_t)2 * G4 * 4);
    float*        hfin = (float*)       carve((size_t)HID * 4);
    unsigned int* cnt  = (unsigned int*)carve((size_t)SEQ * 4);

    // Prep (all independent, ordered only vs consumers by the stream)
    prep_embed<<<SEQ, 256, 0, stream>>>(X, emb, Xe);
    prep_wxt  <<<4 * EMB, 256, 0, stream>>>(W_ii, W_if, W_ig, W_io, WxT);
    prep_wh   <<<4096, 256, 0, stream>>>(W_hi, W_hf, W_hg, W_ho, Whc);
    prep_misc <<<8, 256, 0, stream>>>(b_ii, b_hi, b_if, b_hf, b_ig, b_hg,
                                      b_io, b_ho, bx, cnt);

    // K1: WMMA input projection (4096 tiles, 4 waves/block)
    gx_wmma<<<1024, 128, 0, stream>>>(Xe, WxT, bx, G);

    // K2: serial recurrence, 16 cooperating WGs
    lstm_recurrent<<<16, 256, 0, stream>>>(hidden, cell, Whc, G, gh, hfin, cnt);

    // K3: logits
    logits_kernel<<<NC / 256, 256, 0, stream>>>(hfin, W_out, b_out, out);

    (void)in_sizes; (void)n_in; (void)out_size; (void)ws_size;
}
